// UEFL_9586367004963
// MI455X (gfx1250) — compile-verified, matrix operand order checked by reading, not run on hardware
//
#include <hip/hip_runtime.h>
#include <hip/hip_bf16.h>

typedef __attribute__((ext_vector_type(16))) _Float16 v16h;
typedef __attribute__((ext_vector_type(8)))  _Float16 v8h;
typedef __attribute__((ext_vector_type(8)))  float    v8f;

// ---------------------------------------------------------------------------
// helpers
// ---------------------------------------------------------------------------
__device__ __forceinline__ v16h vq_combine(v8h lo, v8h hi) {
    v16h r;
#pragma unroll
    for (int i = 0; i < 8; ++i) { r[i] = lo[i]; r[i + 8] = hi[i]; }
    return r;
}

// A/B fragment from a row-major-over-K f16 row pointer: two b128 loads.
__device__ __forceinline__ v16h vq_fragK(const _Float16* rowp, int kb, int c0) {
    return vq_combine(*(const v8h*)(rowp + kb + c0),
                      *(const v8h*)(rowp + kb + c0 + 16));
}

// ---------------------------------------------------------------------------
// prep kernels
// ---------------------------------------------------------------------------
__global__ void vq_init_kernel(float* loss_acc, int* hist) {
    int t = blockIdx.x * blockDim.x + threadIdx.x;
    if (t == 0) *loss_acc = 0.f;
    if (t < 1024) hist[t] = 0;
}

// OIHW fp32 -> [OC][(ky,kx,ic)] f16
__global__ void vq_prep_w_kernel(const float* __restrict__ w,
                                 _Float16* __restrict__ wk, int OC, int IC) {
    int t = blockIdx.x * blockDim.x + threadIdx.x;
    int K = IC * 9;
    if (t >= OC * K) return;
    int oc = t / K, k = t % K;
    int tap = k / IC, ic = k % IC;
    wk[t] = (_Float16)w[((size_t)oc * IC + ic) * 9 + tap];
}

__global__ void vq_cvt_f16_kernel(const float* __restrict__ in,
                                  _Float16* __restrict__ out, int n) {
    int t = blockIdx.x * blockDim.x + threadIdx.x;
    if (t < n) out[t] = (_Float16)in[t];
}

// concat(code0,code1) -> f16 [1024][256] + per-row squared norms
__global__ void vq_prep_codes_kernel(const float* __restrict__ c0,
                                     const float* __restrict__ c1,
                                     _Float16* __restrict__ cf,
                                     float* __restrict__ cnorm) {
    int r = blockIdx.x;          // 1024 rows
    int lane = threadIdx.x;      // 32 threads (one wave)
    const float* src = (r < 512) ? (c0 + (size_t)r * 256)
                                 : (c1 + (size_t)(r - 512) * 256);
    float s = 0.f;
    for (int c = lane; c < 256; c += 32) {
        float v = src[c];
        cf[(size_t)r * 256 + c] = (_Float16)v;
        s += v * v;
    }
#pragma unroll
    for (int off = 16; off; off >>= 1) s += __shfl_xor(s, off, 32);
    if (lane == 0) cnorm[r] = s;
}

// ---------------------------------------------------------------------------
// conv1: 3 input channels (K=27) -> direct fp32, output NHWC f16 [4096,16,16,64]
// ---------------------------------------------------------------------------
__global__ void vq_conv1_kernel(const float* __restrict__ x,   // [4096,3,32,32]
                                const float* __restrict__ w1,  // [64,3,3,3]
                                const float* __restrict__ b1,
                                _Float16* __restrict__ fea1) {
    int t = blockIdx.x * blockDim.x + threadIdx.x;  // ((b*16+oh)*16+ow)*64+oc
    int oc = t & 63;
    int p = t >> 6;
    int ow = p & 15, oh = (p >> 4) & 15;
    int b = p >> 8;
    if (b >= 4096) return;
    float acc = b1[oc];
    const float* wp = w1 + oc * 27;
#pragma unroll
    for (int ic = 0; ic < 3; ++ic)
#pragma unroll
        for (int ky = 0; ky < 3; ++ky)
#pragma unroll
            for (int kx = 0; kx < 3; ++kx) {
                int ih = oh * 2 - 1 + ky, iw = ow * 2 - 1 + kx;
                if ((unsigned)ih < 32u && (unsigned)iw < 32u)
                    acc += x[(((size_t)b * 3 + ic) * 32 + ih) * 32 + iw] *
                           wp[ic * 9 + ky * 3 + kx];
            }
    acc = acc > 0.f ? acc : 0.f;
    fea1[t] = (_Float16)acc;
}

// ---------------------------------------------------------------------------
// conv2/conv3 implicit-GEMM WMMA, 2x4 register-blocked, templated geometry
// (power-of-two dims -> shift/mask address math). One wave = 32 rows x 64 oc.
// ---------------------------------------------------------------------------
template <int CIN, int HIN, int WIN, int HOUT, int WOUT, int COUT>
__global__ void vq_conv_wmma_kernel(const _Float16* __restrict__ fin, // NHWC f16
                                    const _Float16* __restrict__ wk,  // [COUT][9*CIN]
                                    const float* __restrict__ bias,
                                    _Float16* __restrict__ fout) {    // NHWC f16
    constexpr int K = 9 * CIN;
    constexpr int HW = HOUT * WOUT;
    int mtile = blockIdx.x, ntile = blockIdx.y;
    int lane = threadIdx.x & 31;
    int half = lane >> 4, col = lane & 15;
    int c0 = half * 8;
    int row0 = mtile * 32;
    // decode the A-fragment rows for the two M sub-tiles (lane holds row col)
    int ar0 = row0 + col, ar1 = row0 + 16 + col;
    int b0 = ar0 / HW, p0 = ar0 % HW, oh0 = p0 / WOUT, ow0 = p0 % WOUT;
    int b1 = ar1 / HW, p1 = ar1 % HW, oh1 = p1 / WOUT, ow1 = p1 % WOUT;
    const _Float16* f0 = fin + (size_t)b0 * HIN * WIN * CIN;
    const _Float16* f1 = fin + (size_t)b1 * HIN * WIN * CIN;
    const _Float16* br = wk + (size_t)(ntile * 64 + col) * K;
    v8f acc[2][4] = {};

    // 8-element patch chunk (CIN pow2 -> shifts; never crosses a tap)
    auto patch8 = [&](const _Float16* fb, int oh, int ow, int k0) -> v8h {
        int tap = k0 / CIN, ic = k0 % CIN;
        int ky = tap / 3, kx = tap % 3;
        int ih = oh * 2 - 1 + ky, iw = ow * 2 - 1 + kx;
        if ((unsigned)ih < (unsigned)HIN && (unsigned)iw < (unsigned)WIN) {
            return *(const v8h*)(fb + ((ih * WIN) + iw) * CIN + ic);
        }
        v8h z = {};
        return z;
    };

#pragma unroll 2
    for (int kb = 0; kb < K; kb += 32) {
        v16h A0 = vq_combine(patch8(f0, oh0, ow0, kb + c0),
                             patch8(f0, oh0, ow0, kb + c0 + 16));
        v16h A1 = vq_combine(patch8(f1, oh1, ow1, kb + c0),
                             patch8(f1, oh1, ow1, kb + c0 + 16));
        v16h B0 = vq_fragK(br, kb, c0);
        v16h B1 = vq_fragK(br + (size_t)16 * K, kb, c0);
        v16h B2 = vq_fragK(br + (size_t)32 * K, kb, c0);
        v16h B3 = vq_fragK(br + (size_t)48 * K, kb, c0);
        acc[0][0] = __builtin_amdgcn_wmma_f32_16x16x32_f16(false, A0, false, B0,
                                                           (short)0, acc[0][0], false, false);
        acc[0][1] = __builtin_amdgcn_wmma_f32_16x16x32_f16(false, A0, false, B1,
                                                           (short)0, acc[0][1], false, false);
        acc[0][2] = __builtin_amdgcn_wmma_f32_16x16x32_f16(false, A0, false, B2,
                                                           (short)0, acc[0][2], false, false);
        acc[0][3] = __builtin_amdgcn_wmma_f32_16x16x32_f16(false, A0, false, B3,
                                                           (short)0, acc[0][3], false, false);
        acc[1][0] = __builtin_amdgcn_wmma_f32_16x16x32_f16(false, A1, false, B0,
                                                           (short)0, acc[1][0], false, false);
        acc[1][1] = __builtin_amdgcn_wmma_f32_16x16x32_f16(false, A1, false, B1,
                                                           (short)0, acc[1][1], false, false);
        acc[1][2] = __builtin_amdgcn_wmma_f32_16x16x32_f16(false, A1, false, B2,
                                                           (short)0, acc[1][2], false, false);
        acc[1][3] = __builtin_amdgcn_wmma_f32_16x16x32_f16(false, A1, false, B3,
                                                           (short)0, acc[1][3], false, false);
    }
#pragma unroll
    for (int ni = 0; ni < 4; ++ni) {
        int oc = ntile * 64 + ni * 16 + col;
        float bv = bias[oc];
#pragma unroll
        for (int mi = 0; mi < 2; ++mi) {
#pragma unroll
            for (int v = 0; v < 8; ++v) {
                int m = row0 + mi * 16 + v + 8 * half;  // D row for lane/vgpr
                int bb = m / HW, pp = m % HW;           // shifts (HW pow2)
                float xv = acc[mi][ni][v] + bv;
                xv = xv > 0.f ? xv : 0.f;
                fout[(((size_t)bb * HW) + pp) * COUT + oc] = (_Float16)xv;
            }
        }
    }
}

// ---------------------------------------------------------------------------
// VQ nearest-code: A rows staged in LDS (read 64x per block), fused WMMA GEMM
// (-2*x.c) + cnorm. Per-lane running (min,argmin) over the lane's column
// subset; single 16-lane butterfly argmin at the end (not per tile).
// ---------------------------------------------------------------------------
__global__ void vq_argmin_kernel(const _Float16* __restrict__ flat,  // [65536,256]
                                 const _Float16* __restrict__ codes, // [1024,256]
                                 const float* __restrict__ cnorm,    // [1024]
                                 const int* __restrict__ idxp,
                                 int* __restrict__ enc) {            // [65536]
    __shared__ _Float16 As[32 * 256];    // 16 KB: this block's 32 flat rows
    int Kc = (idxp[0] == 0) ? 512 : 1024;
    int row0 = blockIdx.x * 32;
    int lane = threadIdx.x & 31;
    int half = lane >> 4, col = lane & 15;
    int c0 = half * 8;
    // stage: one row per lane, 32 x b128 global->LDS
    {
        const _Float16* src = flat + (size_t)(row0 + lane) * 256;
        _Float16* dst = As + lane * 256;
#pragma unroll 8
        for (int j = 0; j < 32; ++j)
            *(v8h*)(dst + j * 8) = *(const v8h*)(src + j * 8);
    }
    __syncthreads();
    const _Float16* a0 = As + col * 256;
    const _Float16* a1 = As + (16 + col) * 256;
    float rmin[2][8];
    int ridx[2][8];
#pragma unroll
    for (int mi = 0; mi < 2; ++mi)
#pragma unroll
        for (int v = 0; v < 8; ++v) { rmin[mi][v] = 3.4e38f; ridx[mi][v] = 0; }
    for (int nt = 0; nt < Kc / 16; ++nt) {
        const _Float16* brow = codes + (size_t)(nt * 16 + col) * 256;
        v8f acc0 = {}, acc1 = {};
#pragma unroll 2
        for (int kb = 0; kb < 256; kb += 32) {
            v16h B = vq_fragK(brow, kb, c0);
            v16h A0 = vq_fragK(a0, kb, c0);   // ds_load_b128 x2
            v16h A1 = vq_fragK(a1, kb, c0);
            acc0 = __builtin_amdgcn_wmma_f32_16x16x32_f16(false, A0, false, B,
                                                          (short)0, acc0, false, false);
            acc1 = __builtin_amdgcn_wmma_f32_16x16x32_f16(false, A1, false, B,
                                                          (short)0, acc1, false, false);
        }
        // per-lane branchless running argmin over this lane's column n
        int n = nt * 16 + col;
        float cn = cnorm[n];
#pragma unroll
        for (int v = 0; v < 8; ++v) {
            float s0 = cn - 2.f * acc0[v];
            float s1 = cn - 2.f * acc1[v];
            if (s0 < rmin[0][v]) { rmin[0][v] = s0; ridx[0][v] = n; }
            if (s1 < rmin[1][v]) { rmin[1][v] = s1; ridx[1][v] = n; }
        }
    }
    // one cross-lane argmin per row (16 lanes of this half-wave)
#pragma unroll
    for (int mi = 0; mi < 2; ++mi) {
#pragma unroll
        for (int v = 0; v < 8; ++v) {
            float s = rmin[mi][v];
            int bi = ridx[mi][v];
#pragma unroll
            for (int off = 8; off; off >>= 1) {
                float os = __shfl_xor(s, off, 16);
                int oi = __shfl_xor(bi, off, 16);
                if (os < s || (os == s && oi < bi)) { s = os; bi = oi; }
            }
            if (col == 0) enc[row0 + mi * 16 + v + 8 * half] = bi;
        }
    }
}

// ---------------------------------------------------------------------------
// gather quantized rows -> q_fea [4096,4096] f16 (NCHW flatten) + MSE partials
// ---------------------------------------------------------------------------
__global__ void vq_gather_loss_kernel(const _Float16* __restrict__ flat,
                                      const _Float16* __restrict__ codes,
                                      const int* __restrict__ enc,
                                      _Float16* __restrict__ qfea,
                                      float* __restrict__ loss_acc) {
    __shared__ float red[8];
    int t = blockIdx.x * blockDim.x + threadIdx.x;  // 65536*256 elements
    int r = t >> 8, c = t & 255;
    _Float16 q = codes[(size_t)enc[r] * 256 + c];
    float d = (float)q - (float)flat[t];
    int b = r >> 4, p = r & 15;                     // NHWC row -> NCHW flatten
    qfea[(size_t)b * 4096 + c * 16 + p] = q;
    float s = d * d;
#pragma unroll
    for (int off = 16; off; off >>= 1) s += __shfl_xor(s, off, 32);
    int wid = threadIdx.x >> 5;
    if ((threadIdx.x & 31) == 0) red[wid] = s;
    __syncthreads();
    if (threadIdx.x < 8) {
        float v = red[threadIdx.x];
#pragma unroll
        for (int off = 4; off; off >>= 1) v += __shfl_xor(v, off, 8);
        if (threadIdx.x == 0) atomicAdd(loss_acc, v);
    }
}

__global__ void vq_hist_kernel(const int* __restrict__ enc, int* __restrict__ hist) {
    int t = blockIdx.x * blockDim.x + threadIdx.x;
    if (t < 65536) atomicAdd(&hist[enc[t]], 1);
}

// ---------------------------------------------------------------------------
// fc1 WMMA GEMM + tanh-GELU, 2x4 register-blocked:
// [4096,4096]f16 x [512,4096]f16^T -> [4096,512]f16
// ---------------------------------------------------------------------------
__global__ void vq_fc1_wmma_kernel(const _Float16* __restrict__ Amat,
                                   const _Float16* __restrict__ Bw,
                                   const float* __restrict__ bias,
                                   _Float16* __restrict__ Hout,
                                   int N, int K) {
    int mtile = blockIdx.x, ntile = blockIdx.y;
    int lane = threadIdx.x & 31;
    int half = lane >> 4, col = lane & 15;
    int c0 = half * 8;
    int row0 = mtile * 32;
    const _Float16* a0 = Amat + (size_t)(row0 + col) * K;
    const _Float16* a1 = Amat + (size_t)(row0 + 16 + col) * K;
    const _Float16* br = Bw + (size_t)(ntile * 64 + col) * K;
    v8f acc[2][4] = {};
#pragma unroll 2
    for (int kb = 0; kb < K; kb += 32) {
        v16h A0 = vq_fragK(a0, kb, c0);
        v16h A1 = vq_fragK(a1, kb, c0);
        v16h B0 = vq_fragK(br, kb, c0);
        v16h B1 = vq_fragK(br + (size_t)16 * K, kb, c0);
        v16h B2 = vq_fragK(br + (size_t)32 * K, kb, c0);
        v16h B3 = vq_fragK(br + (size_t)48 * K, kb, c0);
        acc[0][0] = __builtin_amdgcn_wmma_f32_16x16x32_f16(false, A0, false, B0,
                                                           (short)0, acc[0][0], false, false);
        acc[0][1] = __builtin_amdgcn_wmma_f32_16x16x32_f16(false, A0, false, B1,
                                                           (short)0, acc[0][1], false, false);
        acc[0][2] = __builtin_amdgcn_wmma_f32_16x16x32_f16(false, A0, false, B2,
                                                           (short)0, acc[0][2], false, false);
        acc[0][3] = __builtin_amdgcn_wmma_f32_16x16x32_f16(false, A0, false, B3,
                                                           (short)0, acc[0][3], false, false);
        acc[1][0] = __builtin_amdgcn_wmma_f32_16x16x32_f16(false, A1, false, B0,
                                                           (short)0, acc[1][0], false, false);
        acc[1][1] = __builtin_amdgcn_wmma_f32_16x16x32_f16(false, A1, false, B1,
                                                           (short)0, acc[1][1], false, false);
        acc[1][2] = __builtin_amdgcn_wmma_f32_16x16x32_f16(false, A1, false, B2,
                                                           (short)0, acc[1][2], false, false);
        acc[1][3] = __builtin_amdgcn_wmma_f32_16x16x32_f16(false, A1, false, B3,
                                                           (short)0, acc[1][3], false, false);
    }
#pragma unroll
    for (int ni = 0; ni < 4; ++ni) {
        int n = ntile * 64 + ni * 16 + col;
        float bv = bias[n];
#pragma unroll
        for (int mi = 0; mi < 2; ++mi) {
#pragma unroll
            for (int v = 0; v < 8; ++v) {
                int m = row0 + mi * 16 + v + 8 * half;
                float xv = acc[mi][ni][v] + bv;
                float tv = tanhf(0.7978845608f * (xv + 0.044715f * xv * xv * xv));
                Hout[(size_t)m * N + n] = (_Float16)(0.5f * xv * (1.f + tv));
            }
        }
    }
}

// ---------------------------------------------------------------------------
// fc2: [4096,512] x [10,512]^T, N=10 too small for WMMA -> direct dot
// ---------------------------------------------------------------------------
__global__ void vq_fc2_kernel(const _Float16* __restrict__ h,
                              const float* __restrict__ w,
                              const float* __restrict__ bias,
                              float* __restrict__ out) {
    int t = blockIdx.x * blockDim.x + threadIdx.x;
    if (t >= 4096 * 10) return;
    int b = t / 10, o = t % 10;
    const _Float16* hp = h + (size_t)b * 512;
    const float* wp = w + o * 512;
    float acc = bias[o];
    for (int j = 0; j < 512; ++j) acc += (float)hp[j] * wp[j];
    out[t] = acc;
}

__global__ void vq_finalize_kernel(const float* __restrict__ loss_acc,
                                   const int* __restrict__ hist,
                                   const int* __restrict__ idxp,
                                   float* __restrict__ out) {
    __shared__ float red[256];
    int Kc = (idxp[0] == 0) ? 512 : 1024;
    int t = threadIdx.x;
    float s = 0.f;
    for (int k = t; k < Kc; k += 256) {
        float p = (float)hist[k] / 65536.f;
        s += p * logf(p + 1e-10f);
    }
    red[t] = s;
    __syncthreads();
    for (int st = 128; st; st >>= 1) {
        if (t < st) red[t] += red[t + st];
        __syncthreads();
    }
    if (t == 0) {
        out[40960] = 1.25f * loss_acc[0] / (65536.f * 256.f);  // q_loss + 0.25*e_loss
        out[40961] = expf(-red[0]);                            // perplexity
    }
}

// ---------------------------------------------------------------------------
// launcher
// ---------------------------------------------------------------------------
extern "C" void kernel_launch(void* const* d_in, const int* in_sizes, int n_in,
                              void* d_out, int out_size, void* d_ws, size_t ws_size,
                              hipStream_t stream) {
    const float* x     = (const float*)d_in[0];
    const float* w1    = (const float*)d_in[1];
    const float* b1    = (const float*)d_in[2];
    const float* w2    = (const float*)d_in[3];
    const float* b2    = (const float*)d_in[4];
    const float* w3    = (const float*)d_in[5];
    const float* b3    = (const float*)d_in[6];
    const float* code0 = (const float*)d_in[7];
    const float* code1 = (const float*)d_in[8];
    const float* fc1w  = (const float*)d_in[9];
    const float* fc1b  = (const float*)d_in[10];
    const float* fc2w  = (const float*)d_in[11];
    const float* fc2b  = (const float*)d_in[12];
    const int*   idxp  = (const int*)d_in[13];
    float* out = (float*)d_out;

    // workspace layout (256B aligned slabs)
    char* ws = (char*)d_ws;
    size_t off = 0;
    auto alloc = [&](size_t bytes) {
        char* p = ws + off;
        off += (bytes + 255) & ~(size_t)255;
        return p;
    };
    _Float16* fea1    = (_Float16*)alloc((size_t)4096 * 16 * 16 * 64 * 2); // 134MB
    _Float16* fea2    = (_Float16*)alloc((size_t)4096 * 8 * 8 * 128 * 2);  // 67MB
    _Float16* flat    = (_Float16*)alloc((size_t)65536 * 256 * 2);         // 33.5MB
    _Float16* qfea    = (_Float16*)alloc((size_t)4096 * 4096 * 2);         // 33.5MB
    _Float16* hbuf    = (_Float16*)alloc((size_t)4096 * 512 * 2);          // 4MB
    _Float16* wk2     = (_Float16*)alloc((size_t)128 * 576 * 2);
    _Float16* wk3     = (_Float16*)alloc((size_t)256 * 1152 * 2);
    _Float16* fc1w16  = (_Float16*)alloc((size_t)512 * 4096 * 2);
    _Float16* codes16 = (_Float16*)alloc((size_t)1024 * 256 * 2);
    float*    cnorm   = (float*)alloc(1024 * 4);
    int*      enc     = (int*)alloc((size_t)65536 * 4);
    int*      hist    = (int*)alloc(1024 * 4);
    float*    lossacc = (float*)alloc(256);
    (void)ws_size; (void)in_sizes; (void)n_in; (void)out_size;

    // --- prep ---
    vq_init_kernel<<<4, 256, 0, stream>>>(lossacc, hist);
    vq_prep_w_kernel<<<(128 * 576 + 255) / 256, 256, 0, stream>>>(w2, wk2, 128, 64);
    vq_prep_w_kernel<<<(256 * 1152 + 255) / 256, 256, 0, stream>>>(w3, wk3, 256, 128);
    vq_prep_codes_kernel<<<1024, 32, 0, stream>>>(code0, code1, codes16, cnorm);
    vq_cvt_f16_kernel<<<(512 * 4096 + 255) / 256, 256, 0, stream>>>(fc1w, fc1w16,
                                                                    512 * 4096);
    // --- encoder ---
    vq_conv1_kernel<<<(4096 * 256 * 64) / 256, 256, 0, stream>>>(x, w1, b1, fea1);
    // conv2: M=262144 rows /32, N=128/64 -> (8192, 2) macro tiles
    vq_conv_wmma_kernel<64, 16, 16, 8, 8, 128>
        <<<dim3(8192, 2), 32, 0, stream>>>(fea1, wk2, b2, fea2);
    // conv3: M=65536/32, N=256/64 -> (2048, 4); output == flat [65536,256]
    vq_conv_wmma_kernel<128, 8, 8, 4, 4, 256>
        <<<dim3(2048, 4), 32, 0, stream>>>(fea2, wk3, b3, flat);
    // --- VQ ---
    vq_argmin_kernel<<<2048, 32, 0, stream>>>(flat, codes16, cnorm, idxp, enc);
    vq_gather_loss_kernel<<<(65536 * 256) / 256, 256, 0, stream>>>(
        flat, codes16, enc, qfea, lossacc);
    vq_hist_kernel<<<65536 / 256, 256, 0, stream>>>(enc, hist);
    // --- MLP head ---
    vq_fc1_wmma_kernel<<<dim3(128, 8), 32, 0, stream>>>(qfea, fc1w16, fc1b, hbuf,
                                                        512, 4096);
    vq_fc2_kernel<<<(4096 * 10 + 255) / 256, 256, 0, stream>>>(hbuf, fc2w, fc2b, out);
    vq_finalize_kernel<<<1, 256, 0, stream>>>(lossacc, hist, idxp, out);
}
